// Long2DSCSelfAttention_24318104830232
// MI455X (gfx1250) — matrix operand order repaired and checked by reading, not compiled, not run on hardware
//
#include <hip/hip_runtime.h>

typedef __attribute__((ext_vector_type(16))) __bf16 v16bf;
typedef __attribute__((ext_vector_type(8)))  __bf16 v8bf;
typedef __attribute__((ext_vector_type(8)))  float  v8f;

#define BB    2
#define CC    384
#define HH    12
#define MM    32
#define NN    3137
#define NLOC  3136
#define WW    7
#define W2    49
#define NKEY  442      /* 1 global + 9*49 local */
#define NKEYP 480      /* padded to multiple of 32 */
#define SCALE 0.17677669529663687f
#define NEG_INF (-__builtin_inff())

// ---------------------------------------------------------------- helpers
__device__ __forceinline__ v16bf zero16bf() {
  v16bf z;
  #pragma unroll
  for (int i = 0; i < 16; ++i) z[i] = (__bf16)0.0f;
  return z;
}

// A-matrix 16x32 bf16 lane layout (ISA 7.12.2):
// lane row = lane&15 ; half = lane>>4 ; elem e<8 -> K = half*8+e ; e>=8 -> K = 16+half*8+(e-8)
__device__ __forceinline__ v16bf load_a16x32(const __bf16* base, int rowStride,
                                             int row, int kBase, int half) {
  const __bf16* p0 = base + row * rowStride + kBase + half * 8;
  const __bf16* p1 = base + row * rowStride + kBase + 16 + half * 8;
  v8bf lo = *(const v8bf*)p0;
  v8bf hi = *(const v8bf*)p1;
  return __builtin_shufflevector(lo, hi, 0,1,2,3,4,5,6,7,8,9,10,11,12,13,14,15);
}

// B-matrix 32x16 bf16: source stored as Bt[N][K] (K contiguous).
// lane col N = lane&15 ; elem e<8 -> K = half*8+e ; e>=8 -> K = 16+half*8+(e-8)
__device__ __forceinline__ v16bf load_b32x16(const __bf16* base, int rowStride,
                                             int ncol, int kBase, int half) {
  const __bf16* p0 = base + ncol * rowStride + kBase + half * 8;
  v8bf lo = *(const v8bf*)p0;
  v8bf hi = *(const v8bf*)(p0 + 16);
  return __builtin_shufflevector(lo, hi, 0,1,2,3,4,5,6,7,8,9,10,11,12,13,14,15);
}

// A-tile loaded from f32 LDS (probabilities) with on-the-fly bf16 convert
__device__ __forceinline__ v16bf load_a16x32_f32(const float* base, int rowStride,
                                                 int row, int kBase, int half) {
  const float* p0 = base + row * rowStride + kBase + half * 8;
  const float* p1 = base + row * rowStride + kBase + 16 + half * 8;
  float4 a0 = *(const float4*)p0;
  float4 a1 = *(const float4*)(p0 + 4);
  float4 b0 = *(const float4*)p1;
  float4 b1 = *(const float4*)(p1 + 4);
  v16bf r;
  r[0]=(__bf16)a0.x; r[1]=(__bf16)a0.y; r[2]=(__bf16)a0.z; r[3]=(__bf16)a0.w;
  r[4]=(__bf16)a1.x; r[5]=(__bf16)a1.y; r[6]=(__bf16)a1.z; r[7]=(__bf16)a1.w;
  r[8]=(__bf16)b0.x; r[9]=(__bf16)b0.y; r[10]=(__bf16)b0.z; r[11]=(__bf16)b0.w;
  r[12]=(__bf16)b1.x; r[13]=(__bf16)b1.y; r[14]=(__bf16)b1.z; r[15]=(__bf16)b1.w;
  return r;
}

// ---------------------------------------------------------------- small kernels
__global__ void k_f32_to_bf16(const float* __restrict__ in, __bf16* __restrict__ out, int n) {
  int i = blockIdx.x * 256 + threadIdx.x;
  if (i < n) out[i] = (__bf16)in[i];
}

// Wt[c][k] = (bf16) W[k][c]
__global__ void k_wtrans(const float* __restrict__ W, __bf16* __restrict__ Wt, int K, int Cout) {
  int i = blockIdx.x * 256 + threadIdx.x;
  if (i < K * Cout) {
    int k = i / Cout, c = i - k * Cout;
    Wt[c * K + k] = (__bf16)W[i];
  }
}

__global__ void k_zero_u4(uint4* __restrict__ p, int n) {
  int i = blockIdx.x * 256 + threadIdx.x;
  if (i < n) { uint4 z = {0u,0u,0u,0u}; p[i] = z; }
}

// ---------------------------------------------------------------- WMMA projection GEMM
// one wave -> one 16x16 output tile, K = 384 (12 bf16 WMMA steps)
// mode 0: q = scale*(x[:,1:] @ Wq)   -> qc   bf16 [B][H][64][64pad][32]
// mode 1: kv = x @ Wkv               -> kc/vc bf16 [B][H][64][49][32], kglo/vglo
// mode 2: kvg = x @ Wkvg             -> f32 [B*N][768]
// mode 3: out1 = attn_out @ Wproj+b  -> f32 d_out rows 1..3136 of each batch
__global__ __launch_bounds__(256)
void k_gemm_proj(const __bf16* __restrict__ A, const __bf16* __restrict__ Wt,
                 int mode, int rowTiles, int colTiles, int rowLimit,
                 __bf16* __restrict__ qc, __bf16* __restrict__ kc, __bf16* __restrict__ vc,
                 __bf16* __restrict__ kglo, __bf16* __restrict__ vglo,
                 float* __restrict__ outf, const float* __restrict__ bias) {
  int wave = threadIdx.x >> 5;
  int tileId = blockIdx.x * 8 + wave;
  if (tileId >= rowTiles * colTiles) return;
  int rt = tileId / colTiles;
  int ct = tileId - rt * colTiles;
  int lane = threadIdx.x & 31;
  int half = lane >> 4;
  int l15  = lane & 15;

  // per-lane A row mapping (clamped so loads are always in-bounds; masked after load)
  int r = rt * 16 + l15;
  bool av = (r < rowLimit);
  int aRow;
  if (mode == 0) { int rc = av ? r : 0; int b = rc / NLOC, n = rc - b * NLOC; aRow = b * NN + 1 + n; }
  else aRow = av ? r : 0;

  v8f acc;
  #pragma unroll
  for (int i = 0; i < 8; ++i) acc[i] = 0.0f;
  v16bf za = zero16bf();

  #pragma unroll
  for (int ks = 0; ks < 12; ++ks) {
    v16bf a = load_a16x32(A, CC, aRow, ks * 32, half);  // always safe
    a = av ? a : za;                                    // v_cndmask, no exec churn
    v16bf bm = load_b32x16(Wt, CC, ct * 16 + l15, ks * 32, half);
    acc = __builtin_amdgcn_wmma_f32_16x16x32_bf16(false, a, false, bm,
                                                  (short)0, acc, false, false);
  }

  #pragma unroll
  for (int v = 0; v < 8; ++v) {
    int row = rt * 16 + v + half * 8;
    int col = ct * 16 + l15;
    float val = acc[v];
    if (mode == 0) {
      if (row < rowLimit) {
        int b = row / NLOC, n = row - b * NLOC;
        int hh = col >> 5, m = col & 31;
        int rr = n / 56, cc2 = n - rr * 56;
        int ch = (rr / WW) * 8 + (cc2 / WW);
        int l  = (rr % WW) * WW + (cc2 % WW);
        qc[((((b * HH + hh) * 64 + ch) * 64) + l) * MM + m] = (__bf16)(val * SCALE);
      }
    } else if (mode == 1) {
      if (row < rowLimit) {
        int b = row / NN, t = row - b * NN;
        int kv = (col >= CC);
        int c2 = col & (CC - 1);
        int hh = c2 >> 5, m = c2 & 31;
        __bf16 bv = (__bf16)val;
        if (t == 0) {
          (kv ? vglo : kglo)[(b * HH + hh) * MM + m] = bv;
        } else {
          int n = t - 1;
          int rr = n / 56, cc2 = n - rr * 56;
          int ch = (rr / WW) * 8 + (cc2 / WW);
          int l  = (rr % WW) * WW + (cc2 % WW);
          (kv ? vc : kc)[((((b * HH + hh) * 64 + ch) * W2) + l) * MM + m] = bv;
        }
      }
    } else if (mode == 2) {
      if (row < rowLimit) outf[row * (2 * CC) + col] = val;
    } else {
      if (row < rowLimit) {
        int b = row / NLOC, n = row - b * NLOC;
        outf[(b * NN + 1 + n) * CC + col] = val + bias[col];
      }
    }
  }
}

// ---------------------------------------------------------------- local windowed attention
// one workgroup (8 waves) per (b, h, chunk)
__global__ __launch_bounds__(256)
void k_local_attn(const __bf16* __restrict__ qc, const __bf16* __restrict__ kc,
                  const __bf16* __restrict__ vc, const __bf16* __restrict__ kglo,
                  const __bf16* __restrict__ vglo, const float* __restrict__ rpe,
                  const float* __restrict__ g2l, __bf16* __restrict__ attn_out) {
  __shared__ __attribute__((aligned(16))) __bf16 Qs[64 * MM];        // 4 KB
  __shared__ __attribute__((aligned(16))) __bf16 Ks[NKEYP * MM];     // 30 KB  [key][ch]
  __shared__ __attribute__((aligned(16))) __bf16 Vt[MM * NKEYP];     // 30 KB  [ch][key]
  __shared__ __attribute__((aligned(16))) float  Ss[W2 * NKEYP];     // 94 KB  [l][key]

  int blk   = blockIdx.x;
  int chunk = blk & 63;
  int h     = (blk >> 6) % HH;
  int b     = blk / (64 * HH);
  int cx = chunk >> 3, cy = chunk & 7;
  int tid  = threadIdx.x;
  int lane = tid & 31;
  int wave = tid >> 5;
  int half = lane >> 4;
  int l15  = lane & 15;

  // prefetch center-chunk K/V rows (global_prefetch_b8)
  {
    int coff = ((((b * HH + h) * 64 + chunk) * W2)) * MM;
    __builtin_prefetch(kc + coff, 0, 3);
    __builtin_prefetch(vc + coff, 0, 3);
  }

  // ---- stage Q tile (4 KB) via async global->LDS DMA (ASYNCcnt path)
  {
    const __bf16* qsrc = qc + (size_t)(((b * HH + h) * 64 + chunk) * 64) * MM;
    unsigned ldsaddr = (unsigned)(unsigned long long)(size_t)(&Qs[0]) + (unsigned)tid * 16u;
    unsigned goff    = (unsigned)tid * 16u;
    unsigned long long gbase = (unsigned long long)(size_t)qsrc;
    asm volatile("global_load_async_to_lds_b128 %0, %1, %2\n\t"
                 "s_wait_asynccnt 0x0"
                 :
                 : "v"(ldsaddr), "v"(goff), "s"(gbase)
                 : "memory");
  }
  // ---- stage K rows and transposed V
  for (int kk = tid; kk < NKEYP; kk += 256) {
    uint4 kd[4], vd[4];
    bool have = false;
    const __bf16 *ksrc = nullptr, *vsrc = nullptr;
    if (kk == 0) {
      ksrc = kglo + (b * HH + h) * MM;
      vsrc = vglo + (b * HH + h) * MM;
      have = true;
    } else if (kk < NKEY) {
      int q = kk - 1;
      int nb = q / W2, t = q - nb * W2;
      int mo = cx + nb / 3 - 1, no = cy + nb % 3 - 1;
      if ((unsigned)mo < 8u && (unsigned)no < 8u) {
        int off = ((((b * HH + h) * 64 + (mo * 8 + no)) * W2) + t) * MM;
        ksrc = kc + off; vsrc = vc + off; have = true;
      }
    }
    if (have) {
      #pragma unroll
      for (int i = 0; i < 4; ++i) { kd[i] = ((const uint4*)ksrc)[i]; vd[i] = ((const uint4*)vsrc)[i]; }
    } else {
      uint4 z = {0u,0u,0u,0u};
      #pragma unroll
      for (int i = 0; i < 4; ++i) { kd[i] = z; vd[i] = z; }
    }
    #pragma unroll
    for (int i = 0; i < 4; ++i) ((uint4*)&Ks[kk * MM])[i] = kd[i];
    const __bf16* vp = (const __bf16*)vd;
    #pragma unroll
    for (int m = 0; m < MM; ++m) Vt[m * NKEYP + kk] = vp[m];
  }
  __syncthreads();

  // ---- S = Q K^T + bias / mask   (4 q-tiles x 30 k-tiles, split over 8 waves)
  float gl1 = g2l[HH + h];                         // g2l[1][h][0]
  for (int tile = wave; tile < 4 * (NKEYP / 16); tile += 8) {
    int qt = tile / (NKEYP / 16);
    int kt = tile - qt * (NKEYP / 16);
    v16bf a  = load_a16x32(Qs, MM, qt * 16 + l15, 0, half);
    v16bf bm = load_b32x16(Ks, MM, kt * 16 + l15, 0, half);
    v8f c;
    #pragma unroll
    for (int i = 0; i < 8; ++i) c[i] = 0.0f;
    c = __builtin_amdgcn_wmma_f32_16x16x32_bf16(false, a, false, bm,
                                                (short)0, c, false, false);
    #pragma unroll
    for (int v = 0; v < 8; ++v) {
      int row = qt * 16 + v + half * 8;
      if (row >= W2) continue;
      int col = kt * 16 + l15;
      float sval = c[v];
      if (col == 0) {
        sval += gl1;
      } else if (col < NKEY) {
        int q = col - 1;
        int nb = q / W2, t = q - nb * W2;
        int di = nb / 3, dj = nb - di * 3;
        int mo = cx + di - 1, no = cy + dj - 1;
        if ((unsigned)mo < 8u && (unsigned)no < 8u) {
          int i0 = row / WW - di * WW - t / WW + 20;   // +w + (2w-1)
          int i1 = row % WW - dj * WW - t % WW + 20;
          sval += rpe[(i0 * 27 + i1) * HH + h];
        } else sval = NEG_INF;
      } else sval = NEG_INF;
      Ss[row * NKEYP + col] = sval;
    }
  }
  __syncthreads();

  // ---- softmax over 442 valid cols (padding cols carry -inf -> 0)
  for (int row = wave; row < W2; row += 8) {
    float vals[NKEYP / 32];
    #pragma unroll
    for (int j = 0; j < NKEYP / 32; ++j) vals[j] = Ss[row * NKEYP + j * 32 + lane];
    float mx = NEG_INF;
    #pragma unroll
    for (int j = 0; j < NKEYP / 32; ++j) mx = fmaxf(mx, vals[j]);
    #pragma unroll
    for (int off = 16; off >= 1; off >>= 1) mx = fmaxf(mx, __shfl_xor(mx, off, 32));
    float sum = 0.0f;
    #pragma unroll
    for (int j = 0; j < NKEYP / 32; ++j) { float e = __expf(vals[j] - mx); vals[j] = e; sum += e; }
    #pragma unroll
    for (int off = 16; off >= 1; off >>= 1) sum += __shfl_xor(sum, off, 32);
    float inv = 1.0f / sum;
    #pragma unroll
    for (int j = 0; j < NKEYP / 32; ++j) Ss[row * NKEYP + j * 32 + lane] = vals[j] * inv;
  }
  __syncthreads();

  // ---- X1 = P V   (8 output tiles = 4 row-tiles x 2 channel-tiles, one per wave)
  {
    int qt = wave >> 1, ct = wave & 1;
    int arow = qt * 16 + l15;
    bool rvalid = (arow < W2);
    int arowc = rvalid ? arow : (W2 - 1);            // clamp: load always in-bounds
    v8f acc;
    #pragma unroll
    for (int i = 0; i < 8; ++i) acc[i] = 0.0f;
    v16bf za = zero16bf();
    #pragma unroll
    for (int ks = 0; ks < NKEYP / 32; ++ks) {
      v16bf a = load_a16x32_f32(Ss, NKEYP, arowc, ks * 32, half);
      a = rvalid ? a : za;
      v16bf bm = load_b32x16(Vt, NKEYP, ct * 16 + l15, ks * 32, half);
      acc = __builtin_amdgcn_wmma_f32_16x16x32_bf16(false, a, false, bm,
                                                    (short)0, acc, false, false);
    }
    #pragma unroll
    for (int v = 0; v < 8; ++v) {
      int l = qt * 16 + v + half * 8;
      if (l >= W2) continue;
      int m = ct * 16 + l15;
      int rr = cx * WW + l / WW;
      int cc2 = cy * WW + l % WW;
      int n = rr * 56 + cc2;
      attn_out[(size_t)(b * NLOC + n) * CC + h * MM + m] = (__bf16)acc[v];
    }
  }
}

// ---------------------------------------------------------------- global token attention
__global__ __launch_bounds__(256)
void k_global_attn(const float* __restrict__ x, const float* __restrict__ Wqg,
                   const float* __restrict__ kvg, const float* __restrict__ g2l,
                   const float* __restrict__ g2g, float* __restrict__ x0f) {
  __shared__ float qg[MM];
  __shared__ float logits[NN];
  __shared__ float red[8];
  int b = blockIdx.x / HH, h = blockIdx.x % HH;
  int tid = threadIdx.x, lane = tid & 31, wave = tid >> 5;

  if (tid < MM) {
    const float* xr = x + (size_t)b * NN * CC;     // global token row
    float s = 0.0f;
    for (int k = 0; k < CC; ++k) s += xr[k] * Wqg[k * CC + h * MM + tid];
    qg[tid] = s * SCALE;
  }
  __syncthreads();

  for (int t = tid; t < NN; t += 256) {
    const float* kr = kvg + (size_t)(b * NN + t) * (2 * CC) + h * MM;
    float s = 0.0f;
    #pragma unroll
    for (int m = 0; m < MM; ++m) s += qg[m] * kr[m];
    s += (t == 0) ? g2g[h] : g2l[h];
    logits[t] = s;
  }
  __syncthreads();

  float mx = NEG_INF;
  for (int t = tid; t < NN; t += 256) mx = fmaxf(mx, logits[t]);
  #pragma unroll
  for (int off = 16; off >= 1; off >>= 1) mx = fmaxf(mx, __shfl_xor(mx, off, 32));
  if (lane == 0) red[wave] = mx;
  __syncthreads();
  if (tid == 0) { float m = red[0]; for (int w = 1; w < 8; ++w) m = fmaxf(m, red[w]); red[0] = m; }
  __syncthreads();
  mx = red[0];
  __syncthreads();

  float sum = 0.0f;
  for (int t = tid; t < NN; t += 256) { float e = __expf(logits[t] - mx); logits[t] = e; sum += e; }
  #pragma unroll
  for (int off = 16; off >= 1; off >>= 1) sum += __shfl_xor(sum, off, 32);
  if (lane == 0) red[wave] = sum;
  __syncthreads();
  if (tid == 0) { float s = 0.0f; for (int w = 0; w < 8; ++w) s += red[w]; red[0] = s; }
  __syncthreads();
  float inv = 1.0f / red[0];
  __syncthreads();

  if (tid < MM) {
    float s = 0.0f;
    const float* vbase = kvg + (size_t)b * NN * (2 * CC) + CC + h * MM + tid;
    for (int t = 0; t < NN; ++t) s += logits[t] * vbase[(size_t)t * (2 * CC)];
    x0f[(b * HH + h) * MM + tid] = s * inv;
  }
}

__global__ void k_projg(const float* __restrict__ x0f, const float* __restrict__ Wprojg,
                        const float* __restrict__ bprojg, float* __restrict__ out) {
  int idx = blockIdx.x * 256 + threadIdx.x;
  if (idx < BB * CC) {
    int b = idx / CC, c = idx - b * CC;
    float s = bprojg[c];
    for (int k = 0; k < CC; ++k) s += x0f[b * CC + k] * Wprojg[k * CC + c];
    out[(size_t)(b * NN) * CC + c] = s;   // row 0 of each batch
  }
}

// ---------------------------------------------------------------- launch
extern "C" void kernel_launch(void* const* d_in, const int* in_sizes, int n_in,
                              void* d_out, int out_size, void* d_ws, size_t ws_size,
                              hipStream_t stream) {
  (void)in_sizes; (void)n_in; (void)out_size; (void)ws_size;
  const float* x      = (const float*)d_in[0];
  const float* Wq     = (const float*)d_in[1];
  const float* Wkv    = (const float*)d_in[2];
  const float* Wproj  = (const float*)d_in[3];
  const float* bproj  = (const float*)d_in[4];
  const float* Wqg    = (const float*)d_in[5];
  const float* Wkvg   = (const float*)d_in[6];
  const float* Wprojg = (const float*)d_in[7];
  const float* bprojg = (const float*)d_in[8];
  const float* rpe    = (const float*)d_in[9];
  const float* g2l    = (const float*)d_in[10];
  const float* g2g    = (const float*)d_in[11];
  float* out = (float*)d_out;

  char* ws = (char*)d_ws;
  size_t off = 0;
  auto alloc = [&](size_t bytes) -> void* {
    void* p = ws + off;
    off += (bytes + 255) & ~(size_t)255;
    return p;
  };
  __bf16* xb      = (__bf16*)alloc((size_t)BB * NN * CC * 2);
  __bf16* Wq_t    = (__bf16*)alloc((size_t)CC * CC * 2);
  __bf16* Wkv_t   = (__bf16*)alloc((size_t)CC * 2 * CC * 2);
  __bf16* Wkvg_t  = (__bf16*)alloc((size_t)CC * 2 * CC * 2);
  __bf16* Wproj_t = (__bf16*)alloc((size_t)CC * CC * 2);
  __bf16* qc      = (__bf16*)alloc((size_t)BB * HH * 64 * 64 * MM * 2);
  __bf16* kc      = (__bf16*)alloc((size_t)BB * HH * 64 * W2 * MM * 2);
  __bf16* vc      = (__bf16*)alloc((size_t)BB * HH * 64 * W2 * MM * 2);
  __bf16* kglo    = (__bf16*)alloc((size_t)BB * HH * MM * 2);
  __bf16* vglo    = (__bf16*)alloc((size_t)BB * HH * MM * 2);
  float*  kvg     = (float*)alloc((size_t)BB * NN * 2 * CC * 4);
  __bf16* attn_o  = (__bf16*)alloc((size_t)BB * NLOC * CC * 2);
  float*  x0f     = (float*)alloc((size_t)BB * CC * 4);

  const int nxb = BB * NN * CC;
  k_f32_to_bf16<<<(nxb + 255) / 256, 256, 0, stream>>>(x, xb, nxb);
  k_wtrans<<<(CC * CC + 255) / 256, 256, 0, stream>>>(Wq, Wq_t, CC, CC);
  k_wtrans<<<(CC * 2 * CC + 255) / 256, 256, 0, stream>>>(Wkv, Wkv_t, CC, 2 * CC);
  k_wtrans<<<(CC * 2 * CC + 255) / 256, 256, 0, stream>>>(Wkvg, Wkvg_t, CC, 2 * CC);
  k_wtrans<<<(CC * CC + 255) / 256, 256, 0, stream>>>(Wproj, Wproj_t, CC, CC);

  // zero qc (pad rows 49..63 must be 0)
  {
    int n4 = (int)((size_t)BB * HH * 64 * 64 * MM * 2 / 16);
    k_zero_u4<<<(n4 + 255) / 256, 256, 0, stream>>>((uint4*)qc, n4);
  }

  // mode 0: q projection
  {
    int rT = (BB * NLOC) / 16, cT = CC / 16;
    k_gemm_proj<<<(rT * cT + 7) / 8, 256, 0, stream>>>(
        xb, Wq_t, 0, rT, cT, BB * NLOC, qc, kc, vc, kglo, vglo, nullptr, nullptr);
  }
  // mode 1: kv projection (chunked scatter)
  {
    int rT = (BB * NN + 15) / 16, cT = (2 * CC) / 16;
    k_gemm_proj<<<(rT * cT + 7) / 8, 256, 0, stream>>>(
        xb, Wkv_t, 1, rT, cT, BB * NN, qc, kc, vc, kglo, vglo, nullptr, nullptr);
  }
  // mode 2: kvg projection (f32 flat)
  {
    int rT = (BB * NN + 15) / 16, cT = (2 * CC) / 16;
    k_gemm_proj<<<(rT * cT + 7) / 8, 256, 0, stream>>>(
        xb, Wkvg_t, 2, rT, cT, BB * NN, qc, kc, vc, kglo, vglo, kvg, nullptr);
  }

  // local windowed attention
  k_local_attn<<<BB * HH * 64, 256, 0, stream>>>(qc, kc, vc, kglo, vglo, rpe, g2l, attn_o);

  // mode 3: output projection -> d_out rows 1..NLOC
  {
    int rT = (BB * NLOC) / 16, cT = CC / 16;
    k_gemm_proj<<<(rT * cT + 7) / 8, 256, 0, stream>>>(
        attn_o, Wproj_t, 3, rT, cT, BB * NLOC, qc, kc, vc, kglo, vglo, out, bproj);
  }

  // global token path -> d_out row 0 of each batch
  k_global_attn<<<BB * HH, 256, 0, stream>>>(x, Wqg, kvg, g2l, g2g, x0f);
  k_projg<<<(BB * CC + 255) / 256, 256, 0, stream>>>(x0f, Wprojg, bprojg, out);
}
